// QuantLlamaAttention_57002805952689
// MI455X (gfx1250) — matrix operand order repaired and checked by reading, not compiled, not run on hardware
//
#include <hip/hip_runtime.h>
#include <hip/hip_bf16.h>

// ---------------------------------------------------------------------------
// Llama attention layer for gfx1250 (MI455X), wave32 + WMMA + async-to-LDS.
//  pack W,H -> swizzled bf16 hi/lo  ->  LDS-free bf16x3 WMMA GEMM (QKV)
//  -> RoPE/repack -> flash attention (async K/V staging, online softmax)
//  -> pack attn -> LDS-free bf16x3 WMMA GEMM (O-proj)
// ---------------------------------------------------------------------------

typedef __attribute__((ext_vector_type(16))) __bf16 v16bf;
typedef __attribute__((ext_vector_type(8)))  float  v8f;

union V16U { v16bf v; __bf16 e[16]; uint4 q[2]; };

#define HIDDEN 4096
#define QLEN   2048
#define NHEADS 32
#define HDIM   128

#define WMMA_BF16(A, B, C) \
    __builtin_amdgcn_wmma_f32_16x16x32_bf16(false, (A), false, (B), (short)0, (C), false, false)

// ---------------------------------------------------------------------------
// Pack fp32 matrix into split-bf16 (hi, lo) in WMMA-swizzled layout.
// Layout: [tile][kstep][hi/lo][lane*16 + e], 2KB per (tile,kstep) pair block.
//   A-operand (MxK): lane -> m = tile*16 + lane%16; elem e -> k = kstep*32 +
//                    (lane<16?0:16) + e                          (contiguous, as B)
//   B-operand (KxN): lane -> n = tile*16 + lane%16; elem e -> k = kstep*32 +
//                    (lane<16?0:16) + e                          (contiguous, as A)
// One wave (32 threads) per (tile,kstep) block.
// ---------------------------------------------------------------------------
__global__ __launch_bounds__(256)
void pack_a_swz(const float* __restrict__ A, __bf16* __restrict__ Ap, int M, int K)
{
    int gtid = blockIdx.x * 256 + threadIdx.x;
    int lane = gtid & 31;
    int blk  = gtid >> 5;                 // mtile*KS + kstep
    int KS   = K >> 5;
    int kstep = blk % KS;
    int mtile = blk / KS;
    int m  = mtile * 16 + (lane & 15);
    int kh = (lane < 16) ? 0 : 16;
    V16U hi, lo;
#pragma unroll
    for (int e = 0; e < 16; ++e) {
        int k   = kstep * 32 + kh + e;
        float x = A[(size_t)m * K + k];
        __bf16 hb = (__bf16)x;
        hi.e[e] = hb;
        lo.e[e] = (__bf16)(x - (float)hb);
    }
    __bf16* out = Ap + (size_t)blk * 1024 + lane * 16;
    *(uint4*)(out +   0) = hi.q[0]; *(uint4*)(out +   8) = hi.q[1];
    *(uint4*)(out + 512) = lo.q[0]; *(uint4*)(out + 520) = lo.q[1];
}

__global__ __launch_bounds__(256)
void pack_b_swz(const float* __restrict__ W, __bf16* __restrict__ Wp, int K, int N)
{
    int gtid = blockIdx.x * 256 + threadIdx.x;
    int lane = gtid & 31;
    int blk  = gtid >> 5;                 // ntile*KS + kstep
    int KS   = K >> 5;
    int kstep = blk % KS;
    int ntile = blk / KS;
    int col = ntile * 16 + (lane & 15);
    int kh  = (lane < 16) ? 0 : 16;
    V16U hi, lo;
#pragma unroll
    for (int e = 0; e < 16; ++e) {
        float x = W[(size_t)(kstep * 32 + kh + e) * N + col];
        __bf16 hb = (__bf16)x;
        hi.e[e] = hb;
        lo.e[e] = (__bf16)(x - (float)hb);
    }
    __bf16* out = Wp + (size_t)blk * 1024 + lane * 16;
    *(uint4*)(out +   0) = hi.q[0]; *(uint4*)(out +   8) = hi.q[1];
    *(uint4*)(out + 512) = lo.q[0]; *(uint4*)(out + 520) = lo.q[1];
}

// ---------------------------------------------------------------------------
// C(MxN) = A(MxK) @ B(KxN) from pre-packed split-bf16 operands.
// acc += Ahi*Bhi + Alo*Bhi + Ahi*Blo (fp32 accumulate) ~ fp32 accuracy.
// LDS-free: operands stream straight from L2 in WMMA layout.
// Block = 8 waves = 64(M) x 256(N); each wave owns 32x64 (2 A-tiles x
// 4 B-tiles, 8 accumulators): 12 vector loads vs 24 WMMAs per K-step.
// blockIdx.x = M-block (fastest) so the 32 consecutive blocks share one
// 4.2MB B panel (L2-resident, 192MB L2).
// ---------------------------------------------------------------------------
__global__ __launch_bounds__(256)
void gemm_packed_bf16x3(const __bf16* __restrict__ Ap, const __bf16* __restrict__ Bp,
                        float* __restrict__ C, int M, int N, int K)
{
    const int lane = threadIdx.x & 31;
    const int wave = threadIdx.x >> 5;
    const int KS   = K >> 5;
    const int mt0  = blockIdx.x * 4 + (wave >> 2) * 2;   // first of 2 row tiles
    const int ntb  = blockIdx.y * 16 + (wave & 3) * 4;   // first of 4 col tiles
    const size_t lofs = (size_t)lane * 16;

    const __bf16* a0 = Ap + (size_t)(mt0    ) * KS * 1024 + lofs;
    const __bf16* a1 = Ap + (size_t)(mt0 + 1) * KS * 1024 + lofs;
    const __bf16* b0 = Bp + (size_t)(ntb    ) * KS * 1024 + lofs;
    const __bf16* b1 = Bp + (size_t)(ntb + 1) * KS * 1024 + lofs;
    const __bf16* b2 = Bp + (size_t)(ntb + 2) * KS * 1024 + lofs;
    const __bf16* b3 = Bp + (size_t)(ntb + 3) * KS * 1024 + lofs;

    v8f acc[2][4] = {};

    for (int ks = 0; ks < KS; ++ks) {
        const size_t o = (size_t)ks * 1024;
        v16bf a0h = *(const v16bf*)(a0 + o);
        v16bf a0l = *(const v16bf*)(a0 + o + 512);
        v16bf a1h = *(const v16bf*)(a1 + o);
        v16bf a1l = *(const v16bf*)(a1 + o + 512);
        const __bf16* bb[4] = { b0 + o, b1 + o, b2 + o, b3 + o };
#pragma unroll
        for (int t = 0; t < 4; ++t) {
            v16bf bh = *(const v16bf*)(bb[t]);
            v16bf bl = *(const v16bf*)(bb[t] + 512);
            acc[0][t] = WMMA_BF16(a0h, bh, acc[0][t]);
            acc[0][t] = WMMA_BF16(a0l, bh, acc[0][t]);
            acc[0][t] = WMMA_BF16(a0h, bl, acc[0][t]);
            acc[1][t] = WMMA_BF16(a1h, bh, acc[1][t]);
            acc[1][t] = WMMA_BF16(a1l, bh, acc[1][t]);
            acc[1][t] = WMMA_BF16(a1h, bl, acc[1][t]);
        }
    }

    // C/D layout: vgpr r, lane l -> row = r + 8*(l>=16), col = l%16
    const int rh = (lane >> 4) * 8;
#pragma unroll
    for (int mi = 0; mi < 2; ++mi)
#pragma unroll
        for (int t = 0; t < 4; ++t)
#pragma unroll
            for (int r = 0; r < 8; ++r)
                C[(size_t)((mt0 + mi) * 16 + rh + r) * N + (ntb + t) * 16 + (lane & 15)]
                    = acc[mi][t][r];
}

// ---------------------------------------------------------------------------
// RoPE on Q,K + repack to bf16.  Qb/Kb: [h][s][d]; Vt: [h][d][s] (transposed).
// ---------------------------------------------------------------------------
__global__ __launch_bounds__(256)
void rope_pack(const float* __restrict__ qkv, __bf16* __restrict__ Qb,
               __bf16* __restrict__ Kb, __bf16* __restrict__ Vt)
{
    int idx = blockIdx.x * 256 + threadIdx.x;   // QLEN*NHEADS*64
    int d = idx & 63;
    int h = (idx >> 6) & 31;
    int s = idx >> 11;

    const float* base = qkv + (size_t)s * (3 * HIDDEN) + h * HDIM;
    float fr = (float)s * __expf(-(float)d * 0.14391156831212787f); // 10000^(-d/64)
    float sn, cs;
    __sincosf(fr, &sn, &cs);

    float q0 = base[d],          q1 = base[d + 64];
    float k0 = base[HIDDEN + d], k1 = base[HIDDEN + d + 64];
    size_t o = ((size_t)h * QLEN + s) * HDIM;
    Qb[o + d]      = (__bf16)(q0 * cs - q1 * sn);
    Qb[o + d + 64] = (__bf16)(q1 * cs + q0 * sn);
    Kb[o + d]      = (__bf16)(k0 * cs - k1 * sn);
    Kb[o + d + 64] = (__bf16)(k1 * cs + k0 * sn);

    Vt[((size_t)h * HDIM + d)      * QLEN + s] = (__bf16)base[2 * HIDDEN + d];
    Vt[((size_t)h * HDIM + d + 64) * QLEN + s] = (__bf16)base[2 * HIDDEN + d + 64];
}

// ---------------------------------------------------------------------------
// Causal flash attention.  4 waves/block share head h; wave w owns q-tile
// blockIdx.x*4+w.  K/V 32-key blocks are staged global->LDS with
// GLOBAL_LOAD_ASYNC_TO_LDS_B128 (double buffered, ASYNCcnt discipline).
// Loop count is block-uniform; causally dead blocks are exact no-ops
// (p = exp(-1e30 - m) = 0, corr = 1).
// ---------------------------------------------------------------------------
#define KPAD 136   // 272B row stride: conflict-free 16-lane reads
#define VPAD 40    //  80B row stride

__global__ __launch_bounds__(128)
void flash_attn(const __bf16* __restrict__ Qb, const __bf16* __restrict__ Kb,
                const __bf16* __restrict__ Vt, float* __restrict__ Out)
{
    __shared__ __align__(16) __bf16 Ks[2][32][KPAD];   // 32 keys x 128 dims
    __shared__ __align__(16) __bf16 Vs[2][128][VPAD];  // 128 dims x 32 keys
    __shared__ __bf16 pl[4][16][34];                   // per-wave P staging

    const int tid   = threadIdx.x;
    const int lane  = tid & 31;
    const int w     = tid >> 5;
    const int qtile = blockIdx.x * 4 + w;
    const int h     = blockIdx.y;
    const int m16   = lane & 15;
    const int k0h   = (lane < 16) ? 0 : 16;   // A-layout K base
    const int khB   = (lane < 16) ? 0 : 16;   // B-layout K base
    const int rh    = (lane >> 4) * 8;        // C/D row half

    // stage one 32-key K/V block into buffer `buf` (8 async b128 per thread)
    auto stage = [&](int buf, int kb) {
        const __bf16* gK = Kb + ((size_t)h * QLEN + (size_t)kb * 32) * HDIM;
        const __bf16* gV = Vt + (size_t)h * HDIM * QLEN + (size_t)kb * 32;
#pragma unroll
        for (int i = 0; i < 4; ++i) {
            int sK = tid + i * 128;                  // 512 x 16B segments
            int row = sK >> 4, seg = sK & 15;
            unsigned ldsK = (unsigned)(size_t)&Ks[buf][row][seg * 8];
            unsigned voK  = (unsigned)((row * HDIM + seg * 8) * 2);
            asm volatile("global_load_async_to_lds_b128 %0, %1, %2"
                         :: "v"(ldsK), "v"(voK), "s"(gK) : "memory");
            int sV = tid + i * 128;                  // 512 x 16B segments
            int dd = sV >> 2, vseg = sV & 3;
            unsigned ldsV = (unsigned)(size_t)&Vs[buf][dd][vseg * 8];
            unsigned voV  = (unsigned)((dd * QLEN + vseg * 8) * 2);
            asm volatile("global_load_async_to_lds_b128 %0, %1, %2"
                         :: "v"(ldsV), "v"(voV), "s"(gV) : "memory");
        }
    };

    // Q tile (16x128) in A layout, 4 chunks of K=32
    V16U qa[4];
    const __bf16* qptr = Qb + ((size_t)h * QLEN + qtile * 16 + m16) * HDIM;
#pragma unroll
    for (int c = 0; c < 4; ++c) {
        qa[c].q[0] = *(const uint4*)(qptr + c * 32 + k0h);
        qa[c].q[1] = *(const uint4*)(qptr + c * 32 + k0h + 8);
    }

    v8f   acc[8] = {};
    float mrow[8], lrow[8];
#pragma unroll
    for (int r = 0; r < 8; ++r) { mrow[r] = -3.0e38f; lrow[r] = 0.f; }

    const float scl = 0.08838834764831845f;   // 1/sqrt(128)
    const int   nkb = (((blockIdx.x * 4 + 3) * 16 + 15) >> 5) + 1;  // uniform

    stage(0, 0);

    for (int kb = 0; kb < nkb; ++kb) {
        const int cur = kb & 1;
        if (kb + 1 < nkb) {
            stage(cur ^ 1, kb + 1);
            asm volatile("s_wait_asynccnt 8" ::: "memory");  // cur buffer done
        } else {
            asm volatile("s_wait_asynccnt 0" ::: "memory");
        }
        __syncthreads();

        // ---- scores: two 16x16 subtiles, contraction over d=128
        v8f s0 = {}, s1 = {};
#pragma unroll
        for (int c = 0; c < 4; ++c) {
            V16U kv0, kv1;
            kv0.q[0] = *(const uint4*)&Ks[cur][m16][c * 32 + khB];
            kv0.q[1] = *(const uint4*)&Ks[cur][m16][c * 32 + khB + 8];
            kv1.q[0] = *(const uint4*)&Ks[cur][16 + m16][c * 32 + khB];
            kv1.q[1] = *(const uint4*)&Ks[cur][16 + m16][c * 32 + khB + 8];
            s0 = WMMA_BF16(qa[c].v, kv0.v, s0);
            s1 = WMMA_BF16(qa[c].v, kv1.v, s1);
        }

        // ---- online softmax per row (rows live in 16-lane groups)
#pragma unroll
        for (int r = 0; r < 8; ++r) {
            int   q  = qtile * 16 + rh + r;
            float a0 = s0[r] * scl + ((kb * 32      + m16 <= q) ? 0.f : -1.0e30f);
            float a1 = s1[r] * scl + ((kb * 32 + 16 + m16 <= q) ? 0.f : -1.0e30f);
            float mx = fmaxf(a0, a1);
#pragma unroll
            for (int off = 8; off >= 1; off >>= 1)
                mx = fmaxf(mx, __shfl_xor(mx, off, 16));
            float mnew = fmaxf(mrow[r], mx);
            float corr = __expf(mrow[r] - mnew);
            float p0   = __expf(a0 - mnew);
            float p1   = __expf(a1 - mnew);
            float ps   = p0 + p1;
#pragma unroll
            for (int off = 8; off >= 1; off >>= 1)
                ps += __shfl_xor(ps, off, 16);
            lrow[r] = lrow[r] * corr + ps;
            mrow[r] = mnew;
#pragma unroll
            for (int t = 0; t < 8; ++t) acc[t][r] *= corr;
            pl[w][rh + r][m16]      = (__bf16)p0;      // P row-major 16x32
            pl[w][rh + r][16 + m16] = (__bf16)p1;
        }
        __builtin_amdgcn_wave_barrier();

        // ---- reload P in WMMA A layout (same-wave LDS ops are in-order)
        V16U pa;
#pragma unroll
        for (int e = 0; e < 8; ++e)  pa.e[e] = pl[w][m16][k0h + e];
#pragma unroll
        for (int e = 8; e < 16; ++e) pa.e[e] = pl[w][m16][k0h + e];
        __builtin_amdgcn_wave_barrier();

        // ---- PV: 8 d-subtiles from LDS-staged transposed V
#pragma unroll
        for (int t = 0; t < 8; ++t) {
            V16U vv;
            vv.q[0] = *(const uint4*)&Vs[cur][t * 16 + m16][khB];
            vv.q[1] = *(const uint4*)&Vs[cur][t * 16 + m16][khB + 8];
            acc[t] = WMMA_BF16(pa.v, vv.v, acc[t]);
        }
        __syncthreads();
    }

    // ---- normalize + scatter to [s][hidden] fp32 for the O-projection
#pragma unroll
    for (int t = 0; t < 8; ++t)
#pragma unroll
        for (int r = 0; r < 8; ++r)
            Out[(size_t)(qtile * 16 + rh + r) * HIDDEN + h * HDIM + t * 16 + m16]
                = acc[t][r] / lrow[r];
}

// ---------------------------------------------------------------------------
// Launch.  ws layout (~453 MB):
//   [0,        201326592)  Wqkv packed hi/lo (4096x12288 x2 bf16)
//   [201326592,268435456)  Wo   packed hi/lo (4096x4096  x2 bf16)
//   [268435456,301989888)  Hp packed  -- reused as attn fp32 (2048x4096)
//   [301989888,402653184)  qkv fp32   -- reused as packed attn
//   [402653184,419430400)  Qb bf16 [32][2048][128]
//   [419430400,436207616)  Kb bf16
//   [436207616,452984832)  Vt bf16 [32][128][2048]
// ---------------------------------------------------------------------------
extern "C" void kernel_launch(void* const* d_in, const int* in_sizes, int n_in,
                              void* d_out, int out_size, void* d_ws, size_t ws_size,
                              hipStream_t stream)
{
    (void)in_sizes; (void)n_in; (void)out_size; (void)ws_size;
    const float* hs    = (const float*)d_in[0];
    const float* qkv_w = (const float*)d_in[1];
    const float* o_w   = (const float*)d_in[2];
    // d_in[3] (attention_mask) unused: causal mask computed inline.

    char*   ws    = (char*)d_ws;
    __bf16* Wqkvp = (__bf16*)(ws);
    __bf16* Wop   = (__bf16*)(ws + 201326592u);
    __bf16* Hp    = (__bf16*)(ws + 268435456u);
    float*  attn  = (float*) (ws + 268435456u);   // reuse after QKV GEMM
    float*  qkv   = (float*) (ws + 301989888u);
    __bf16* Ap2   = (__bf16*)(ws + 301989888u);   // reuse after rope_pack
    __bf16* Qb    = (__bf16*)(ws + 402653184u);
    __bf16* Kb    = (__bf16*)(ws + 419430400u);
    __bf16* Vt    = (__bf16*)(ws + 436207616u);

    // one-time operand packing (swizzled split-bf16)
    pack_b_swz<<<12288, 256, 0, stream>>>(qkv_w, Wqkvp, HIDDEN, 3 * HIDDEN);
    pack_b_swz<<< 4096, 256, 0, stream>>>(o_w,   Wop,   HIDDEN, HIDDEN);
    pack_a_swz<<< 2048, 256, 0, stream>>>(hs,    Hp,    QLEN,   HIDDEN);

    // QKV projection
    gemm_packed_bf16x3<<<dim3(QLEN / 64, 3 * HIDDEN / 256), 256, 0, stream>>>(
        Hp, Wqkvp, qkv, QLEN, 3 * HIDDEN, HIDDEN);

    rope_pack<<<(QLEN * NHEADS * 64) / 256, 256, 0, stream>>>(qkv, Qb, Kb, Vt);

    flash_attn<<<dim3(QLEN / 64, NHEADS), 128, 0, stream>>>(Qb, Kb, Vt, attn);

    // O projection
    pack_a_swz<<<2048, 256, 0, stream>>>(attn, Ap2, QLEN, HIDDEN);
    gemm_packed_bf16x3<<<dim3(QLEN / 64, HIDDEN / 256), 256, 0, stream>>>(
        Ap2, Wop, (float*)d_out, QLEN, HIDDEN, HIDDEN);
}
